// SparseExpert_24713241821333
// MI455X (gfx1250) — matrix operand order repaired and checked by reading, not compiled, not run on hardware
//
#include <hip/hip_runtime.h>
#include <math.h>

// ---------------------------------------------------------------------------
// Types for WMMA (CDNA5 / gfx1250, wave32)
// ---------------------------------------------------------------------------
typedef __bf16 v16bf __attribute__((ext_vector_type(16)));
typedef float  v8f   __attribute__((ext_vector_type(8)));
typedef unsigned int u32x4 __attribute__((ext_vector_type(4)));

union FragBF {
  v16bf v;
  u32x4 q[2];
};

__device__ inline unsigned short f2bf(float f) {
  unsigned u = __builtin_bit_cast(unsigned, f);
  u += 0x7FFFu + ((u >> 16) & 1u);   // round-to-nearest-even
  return (unsigned short)(u >> 16);
}
__device__ inline float bf2f(unsigned short h) {
  unsigned u = ((unsigned)h) << 16;
  return __builtin_bit_cast(float, u);
}
// Order-preserving key for bf16 bit patterns (ascending float order)
__device__ inline unsigned bf16key(unsigned short h) {
  return (h & 0x8000u) ? (unsigned)(~h & 0xFFFFu) : (unsigned)(h | 0x8000u);
}

// ---------------------------------------------------------------------------
// Elementwise fp32 -> bf16
// ---------------------------------------------------------------------------
__global__ void k_cvt_bf16(const float* __restrict__ in,
                           unsigned short* __restrict__ out, int n) {
  int i = blockIdx.x * 256 + threadIdx.x;
  if (i < n) out[i] = f2bf(in[i]);
}

// W [K][N] fp32 -> Wt [N][K] bf16, LDS-tiled 32x32 (coalesced on both sides)
__global__ __launch_bounds__(256) void k_transpose_bf16(
    const float* __restrict__ in, unsigned short* __restrict__ out,
    int K, int N) {
  __shared__ float tile[32][33];
  const int kk0 = blockIdx.y * 32;   // K tile origin
  const int n0  = blockIdx.x * 32;   // N tile origin
  const int tx = threadIdx.x & 31;
  const int ty = threadIdx.x >> 5;   // 8 row-groups
#pragma unroll
  for (int r = ty; r < 32; r += 8)
    tile[r][tx] = in[(long long)(kk0 + r) * N + n0 + tx];
  __syncthreads();
#pragma unroll
  for (int r = ty; r < 32; r += 8)
    out[(long long)(n0 + r) * K + kk0 + tx] = f2bf(tile[tx][r]);
}

// ---------------------------------------------------------------------------
// WMMA GEMM: C[M][N] = act(A[M][K] * Bt[N][K]^T + bias)
// 256 threads = 8 waves (2 along M x 4 along N); wave tile 64x64
// (4x4 of 16x16x32 bf16 WMMA); workgroup tile 128x256.
// 16 b128 loads : 16 WMMAs per k-step -> ~32 FLOP per cache byte.
// MODE: 0=ReLU->bf16, 1=none->bf16, 2=GELU(exact)->bf16, 3=none->fp32
// ---------------------------------------------------------------------------
template <int MODE>
__global__ __launch_bounds__(256) void k_gemm_bf16(
    const unsigned short* __restrict__ A,   // [M][K] bf16 bits
    const unsigned short* __restrict__ Bt,  // [N][K] bf16 bits
    const float* __restrict__ bias,         // [N]
    void* __restrict__ outv,
    int M, int N, int K) {
  const int lane  = threadIdx.x & 31;
  const int wave  = threadIdx.x >> 5;
  const int hs    = lane >> 4;    // half-wave select
  const int l16   = lane & 15;
  const int waveN = wave & 3;     // 4 waves across N -> 256 cols
  const int waveM = wave >> 2;    // 2 waves across M -> 128 rows
  const int rowBase = blockIdx.y * 128 + waveM * 64;
  const int colBase = blockIdx.x * 256 + waveN * 64;

  const unsigned short* ap[4];
  const unsigned short* bp[4];
#pragma unroll
  for (int i = 0; i < 4; i++) {
    ap[i] = A  + (size_t)(rowBase + i * 16 + l16) * K;
    bp[i] = Bt + (size_t)(colBase + i * 16 + l16) * K;
  }

  v8f acc[4][4] = {};

  for (int k = 0; k < K; k += 32) {
    FragBF a[4], b[4];
#pragma unroll
    for (int i = 0; i < 4; i++) {
      // A 16x32 bf16: lane<16 holds K {0..7,16..23}, lane>=16 holds {8..15,24..31}
      a[i].q[0] = *(const u32x4*)(ap[i] + k + hs * 8);
      a[i].q[1] = *(const u32x4*)(ap[i] + k + 16 + hs * 8);
      // B 32x16 bf16: lane<16 holds K {0..7,16..23} of col l16, lane>=16 holds {8..15,24..31}
      b[i].q[0] = *(const u32x4*)(bp[i] + k + hs * 8);
      b[i].q[1] = *(const u32x4*)(bp[i] + k + 16 + hs * 8);
    }
    if (k + 512 < K) {
      // L2 prefetch of the streaming A operand ~16 k-steps ahead
#pragma unroll
      for (int i = 0; i < 4; i++) __builtin_prefetch(ap[i] + k + 512, 0, 0);
    }
#pragma unroll
    for (int i = 0; i < 4; i++)
#pragma unroll
      for (int j = 0; j < 4; j++)
        acc[i][j] = __builtin_amdgcn_wmma_f32_16x16x32_bf16(
            false, a[i].v, false, b[j].v, (short)0, acc[i][j], false, false);
  }

  float* outF = (float*)outv;
  unsigned short* outH = (unsigned short*)outv;
#pragma unroll
  for (int i = 0; i < 4; i++) {
#pragma unroll
    for (int j = 0; j < 4; j++) {
      int col = colBase + j * 16 + l16;
      float bv = bias[col];
#pragma unroll
      for (int r = 0; r < 8; r++) {
        // C/D layout: lanes 0-15 hold M=r, lanes 16-31 hold M=r+8; N = lane&15
        int row = rowBase + i * 16 + hs * 8 + r;
        float v = acc[i][j][r] + bv;
        if (MODE == 0) v = v > 0.f ? v : 0.f;
        if (MODE == 2) v = 0.5f * v * (1.f + erff(v * 0.70710678118654752f));
        size_t oi = (size_t)row * N + col;
        if (MODE == 3) outF[oi] = v;
        else           outH[oi] = f2bf(v);
      }
    }
  }
}

// ---------------------------------------------------------------------------
// Gate head: score[row] = sigmoid(dot(hg[row], Wg2) + bg2); one wave per row
// ---------------------------------------------------------------------------
__global__ void k_gate(const unsigned short* __restrict__ hg,
                       const float* __restrict__ Wg2,
                       const float* __restrict__ bg2,
                       float* __restrict__ gate_out, int rows, int Kg) {
  int lane = threadIdx.x & 31;
  int wave = threadIdx.x >> 5;
  int row  = blockIdx.x * 8 + wave;
  if (row >= rows) return;
  const unsigned short* p = hg + (size_t)row * Kg;
  float s = 0.f;
  for (int t = lane; t < Kg; t += 32) s += bf2f(p[t]) * Wg2[t];
#pragma unroll
  for (int off = 16; off > 0; off >>= 1) s += __shfl_xor(s, off, 32);
  if (lane == 0) gate_out[row] = 1.f / (1.f + expf(-(s + bg2[0])));
}

// ---------------------------------------------------------------------------
// Two-level exact kth-smallest over bf16 logits (order keys, 8+8 bits).
// Pass 1: LDS histogram of key>>8, merged into 256 global bins.
// Pass 2: LDS histogram of key&255 for elements inside the coarse bin.
// ---------------------------------------------------------------------------
__global__ void k_hist_init(unsigned* __restrict__ hist /*512*/,
                            unsigned* __restrict__ coarse /*2*/,
                            unsigned* __restrict__ thrkey) {
  int i = blockIdx.x * 256 + threadIdx.x;
  if (i < 512) hist[i] = 0u;
  if (i == 0) { coarse[0] = 0u; coarse[1] = 0u; *thrkey = 0u; }
}

__global__ __launch_bounds__(256) void k_hist_coarse(
    const unsigned short* __restrict__ z, unsigned* __restrict__ hist,
    long long n) {
  __shared__ unsigned h[256];
  h[threadIdx.x] = 0u;
  __syncthreads();
  long long i = (long long)blockIdx.x * blockDim.x + threadIdx.x;
  long long stride = (long long)gridDim.x * blockDim.x;
  for (; i < n; i += stride) atomicAdd(&h[bf16key(z[i]) >> 8], 1u);
  __syncthreads();
  unsigned v = h[threadIdx.x];
  if (v) atomicAdd(&hist[threadIdx.x], v);
}

__global__ void k_select_coarse(const unsigned* __restrict__ hist,
                                unsigned* __restrict__ coarse, long long kth) {
  if (threadIdx.x == 0 && blockIdx.x == 0) {
    unsigned long long c = 0;
    for (int b = 0; b < 256; b++) {
      unsigned long long nc = c + hist[b];
      if (nc >= (unsigned long long)kth) {
        coarse[0] = (unsigned)b;
        coarse[1] = (unsigned)((unsigned long long)kth - c);  // rank inside bin
        break;
      }
      c = nc;
    }
  }
}

__global__ __launch_bounds__(256) void k_hist_fine(
    const unsigned short* __restrict__ z, const unsigned* __restrict__ coarse,
    unsigned* __restrict__ hist2, long long n) {
  __shared__ unsigned h[256];
  h[threadIdx.x] = 0u;
  __syncthreads();
  unsigned cb = coarse[0];
  long long i = (long long)blockIdx.x * blockDim.x + threadIdx.x;
  long long stride = (long long)gridDim.x * blockDim.x;
  for (; i < n; i += stride) {
    unsigned key = bf16key(z[i]);
    if ((key >> 8) == cb) atomicAdd(&h[key & 255u], 1u);
  }
  __syncthreads();
  unsigned v = h[threadIdx.x];
  if (v) atomicAdd(&hist2[threadIdx.x], v);
}

__global__ void k_select_fine(const unsigned* __restrict__ hist2,
                              const unsigned* __restrict__ coarse,
                              unsigned* __restrict__ thrkey) {
  if (threadIdx.x == 0 && blockIdx.x == 0) {
    unsigned rem = coarse[1];
    unsigned c = 0;
    for (int b = 0; b < 256; b++) {
      c += hist2[b];
      if (c >= rem) { *thrkey = (coarse[0] << 8) | (unsigned)b; break; }
    }
  }
}

// ---------------------------------------------------------------------------
// Mask write (0/1 floats) + per-row "any" + per-token scale
// ---------------------------------------------------------------------------
__global__ __launch_bounds__(256) void k_mask(const unsigned short* __restrict__ z,
                       const unsigned* __restrict__ thrkey,
                       float* __restrict__ maskOut,
                       const float* __restrict__ gate,
                       float* __restrict__ scale, int H) {
  __shared__ int s_any;
  if (threadIdx.x == 0) s_any = 0;
  __syncthreads();
  int row = blockIdx.x;
  unsigned tk = *thrkey;
  const unsigned short* p = z + (size_t)row * H;
  float* o = maskOut + (size_t)row * H;
  int any = 0;
  for (int c = threadIdx.x; c < H; c += 256) {
    int m = bf16key(p[c]) > tk;   // strict >, matches imp > threshold
    o[c] = (float)m;
    any |= m;
  }
  if (any) s_any = 1;
  __syncthreads();
  if (threadIdx.x == 0)
    scale[row] = (gate[row] > 0.5f && s_any) ? 1.f : 0.f;
}

__global__ void k_scale(float* __restrict__ out, const float* __restrict__ scale,
                        int n, int O) {
  int i = blockIdx.x * 256 + threadIdx.x;
  if (i < n) out[i] *= scale[i / O];
}

// ---------------------------------------------------------------------------
// Launch
// ---------------------------------------------------------------------------
extern "C" void kernel_launch(void* const* d_in, const int* in_sizes, int n_in,
                              void* d_out, int out_size, void* d_ws, size_t ws_size,
                              hipStream_t stream) {
  const float* x   = (const float*)d_in[0];
  const float* W1  = (const float*)d_in[1];
  const float* b1  = (const float*)d_in[2];
  const float* W2  = (const float*)d_in[3];
  const float* b2  = (const float*)d_in[4];
  const float* Wm1 = (const float*)d_in[5];
  const float* bm1 = (const float*)d_in[6];
  const float* Wm2 = (const float*)d_in[7];
  const float* bm2 = (const float*)d_in[8];
  const float* Wg1 = (const float*)d_in[9];
  const float* bg1 = (const float*)d_in[10];
  const float* Wg2 = (const float*)d_in[11];
  const float* bg2 = (const float*)d_in[12];

  const int Nr = 8192, D = 1024, H = 4096, O = 1024, Hm = 1024, Hg = 512;

  // workspace carve-out
  char* ws = (char*)d_ws;
  size_t off = 0;
  auto walloc = [&](size_t bytes) -> char* {
    char* p = ws + off;
    off += (bytes + 255) & ~(size_t)255;
    return p;
  };
  unsigned short* xb   = (unsigned short*)walloc((size_t)Nr * D * 2);
  unsigned short* W1t  = (unsigned short*)walloc((size_t)H * D * 2);
  unsigned short* W2t  = (unsigned short*)walloc((size_t)O * H * 2);
  unsigned short* Wm1t = (unsigned short*)walloc((size_t)Hm * D * 2);
  unsigned short* Wm2t = (unsigned short*)walloc((size_t)H * Hm * 2);
  unsigned short* Wg1t = (unsigned short*)walloc((size_t)Hg * D * 2);
  unsigned short* hm   = (unsigned short*)walloc((size_t)Nr * Hm * 2);
  unsigned short* zm   = (unsigned short*)walloc((size_t)Nr * H * 2);
  unsigned short* hg   = (unsigned short*)walloc((size_t)Nr * Hg * 2);
  unsigned short* h1   = (unsigned short*)walloc((size_t)Nr * H * 2);
  unsigned*       hist = (unsigned*)walloc(512 * 4);   // [0..255] coarse, [256..511] fine
  unsigned*       crs  = (unsigned*)walloc(256);       // {coarse bin, rank-in-bin}
  unsigned*       thr  = (unsigned*)walloc(256);
  float*          scal = (float*)walloc((size_t)Nr * 4);

  float* out0 = (float*)d_out;            // expert output [Nr*O]
  float* out1 = out0 + (size_t)Nr * O;    // compute_mask as 0/1 floats [Nr*H]
  float* out2 = out1 + (size_t)Nr * H;    // gate_score [Nr]

  // 1) precision conversion / weight transpose (LDS tiled)
  k_cvt_bf16<<<(Nr * D + 255) / 256, 256, 0, stream>>>(x, xb, Nr * D);
  k_transpose_bf16<<<dim3(H / 32, D / 32),  256, 0, stream>>>(W1, W1t, D, H);
  k_transpose_bf16<<<dim3(O / 32, H / 32),  256, 0, stream>>>(W2, W2t, H, O);
  k_transpose_bf16<<<dim3(Hm / 32, D / 32), 256, 0, stream>>>(Wm1, Wm1t, D, Hm);
  k_transpose_bf16<<<dim3(H / 32, Hm / 32), 256, 0, stream>>>(Wm2, Wm2t, Hm, H);
  k_transpose_bf16<<<dim3(Hg / 32, D / 32), 256, 0, stream>>>(Wg1, Wg1t, D, Hg);

  // 2) mask network
  k_gemm_bf16<0><<<dim3(Hm / 256, Nr / 128), 256, 0, stream>>>(xb, Wm1t, bm1, hm, Nr, Hm, D);
  k_gemm_bf16<1><<<dim3(H / 256, Nr / 128), 256, 0, stream>>>(hm, Wm2t, bm2, zm, Nr, H, Hm);

  // 3) gate network
  k_gemm_bf16<0><<<dim3(Hg / 256, Nr / 128), 256, 0, stream>>>(xb, Wg1t, bg1, hg, Nr, Hg, D);
  k_gate<<<Nr / 8, 256, 0, stream>>>(hg, Wg2, bg2, out2, Nr, Hg);

  // 4) kth-value threshold on logits (sigmoid is monotone): two-level select
  const long long nz = (long long)Nr * H;
  k_hist_init<<<2, 256, 0, stream>>>(hist, crs, thr);
  k_hist_coarse<<<1024, 256, 0, stream>>>(zm, hist, nz);
  k_select_coarse<<<1, 32, 0, stream>>>(hist, crs, nz / 2);
  k_hist_fine<<<1024, 256, 0, stream>>>(zm, crs, hist + 256, nz);
  k_select_fine<<<1, 32, 0, stream>>>(hist + 256, crs, thr);

  // 5) expert network
  k_gemm_bf16<2><<<dim3(H / 256, Nr / 128), 256, 0, stream>>>(xb, W1t, b1, h1, Nr, H, D);
  k_gemm_bf16<3><<<dim3(O / 256, Nr / 128), 256, 0, stream>>>(h1, W2t, b2, out0, Nr, O, H);

  // 6) mask output + per-token scale, then apply
  k_mask<<<Nr, 256, 0, stream>>>(zm, thr, out1, out2, scal, H);
  k_scale<<<(Nr * O + 255) / 256, 256, 0, stream>>>(out0, scal, Nr * O, O);
}
